// DetailTransformer_20040317403492
// MI455X (gfx1250) — matrix operand order repaired and checked by reading, not compiled, not run on hardware
//
#include <hip/hip_runtime.h>
#include <hip/hip_bf16.h>
#include <math.h>

// ---------------------------------------------------------------------------
// Problem constants (fixed by the reference)
// ---------------------------------------------------------------------------
#define DOF   69
#define NL    6
#define NH    8
#define DMODEL 512
#define DFF   2048
#define HD    64
#define BB    8
#define TT    512
#define MT    (BB * TT)        // 4096 token rows
#define KIN   (2 * DOF)        // 138
#define RREL  (2 * TT - 1)     // 1023
#define DOUT  (DOF + 4)        // 73
#define ZBH   (BB * NH)        // 64 attention batches

typedef __attribute__((ext_vector_type(16))) _Float16 v16h;
typedef __attribute__((ext_vector_type(8)))  _Float16 v8h;
typedef __attribute__((ext_vector_type(8)))  float    v8f;

// ---------------------------------------------------------------------------
// WMMA GEMM:  C[m][n] = epi( sum_k A[m][k] * Bt[n][k] )
//   BL  = 0 : Bsrc stored N x K (direct), 1 : stored K x N (transpose in LDS)
//   FAST= 1 : M%128==0, N%64==0, K%32==0, leading dims 16B-aligned
//   F16 = 1 : A and B are _Float16 in global
//             (FAST+BL0+F16 uses global_load_async_to_lds_b128 / ASYNCcnt)
// Epilogues (EPI):
//   0 PLAIN f32, 1 BIAS f32, 2 BIAS_PRELU f32, 3 BIAS_RES f32,
//   4 Q/K head scatter (f16 out), 5 V transposed scatter (f16 out),
//   6 SKEW accumulate (f32 RMW), 7 attn@V head gather (f32 out),
//   8 BIAS f16 out
// Block: 256 threads = 8 wave32; tile 128(M) x 64(N) x 32(K), double-buffered.
// ---------------------------------------------------------------------------
#define BM 128
#define BN 64
#define BKK 32
#define LDSA 40   // 32 + 8 halves padding
#define LDSB 40

template<int BL, int EPI, int FAST, int F16>
__global__ __launch_bounds__(256)
void gemm_wmma(const void* __restrict__ Ap, const void* __restrict__ Bp,
               void* __restrict__ Cp,
               int M, int N, int K, int lda, int ldb, int ldc,
               long long sA, long long sB, long long sC,
               const float* __restrict__ bias,
               const float* __restrict__ alpha,
               const float* __restrict__ res, int ldres)
{
    __shared__ __align__(16) _Float16 As[2][BM * LDSA];
    __shared__ __align__(16) _Float16 Bs[2][BN * LDSB];

    const int z  = blockIdx.z;
    const int m0 = blockIdx.y * BM;
    const int n0 = blockIdx.x * BN;

    const float*    A32 = (const float*)Ap    + (F16 ? 0 : (long long)z * sA);
    const float*    B32 = (const float*)Bp    + (F16 ? 0 : (long long)z * sB);
    const _Float16* A16 = (const _Float16*)Ap + (F16 ? (long long)z * sA : 0);
    const _Float16* B16 = (const _Float16*)Bp + (F16 ? (long long)z * sB : 0);
    float*    Cf = (float*)Cp    + (long long)z * sC;
    _Float16* Ch = (_Float16*)Cp + (long long)z * sC;   // f16 outputs use same stride

    const int tid  = threadIdx.x;
    const int wave = tid >> 5;          // 0..7, owns rows [wave*16, wave*16+16)
    const int lane = tid & 31;
    const int lr   = lane & 15;

    // cooperative-load coordinates
    const int ar = tid >> 1;            // A row within tile (0..127)
    const int ac = (tid & 1) * 16;      // A col base (0 or 16)
    const int br = tid >> 2;            // B (BL=0): n row (0..63)
    const int bc = (tid & 3) * 8;       // B (BL=0): k base (0/8/16/24)
    const int bk = tid >> 3;            // B (BL=1): k row (0..31)
    const int bn = (tid & 7) * 8;       // B (BL=1): n base (0..56)

    const bool ASYNC = (F16 && FAST && BL == 0);   // pure-copy tiles -> async DMA

    float aF[16], bF[8];
    _Float16 aH[16], bH[8];

    // ---- async copy issue (F16 + FAST + BL0): global -> LDS, no VGPR staging
    auto asyncA = [&](int k0, int buf) {
        unsigned l0 = (unsigned)(size_t)&As[buf][ar * LDSA + ac];
        unsigned long long g0 =
            (unsigned long long)(size_t)(A16 + (size_t)(m0 + ar) * lda + k0 + ac);
        asm volatile("global_load_async_to_lds_b128 %0, %1, off"
                     :: "v"(l0), "v"(g0) : "memory");
        asm volatile("global_load_async_to_lds_b128 %0, %1, off offset:16"
                     :: "v"(l0), "v"(g0) : "memory");
    };
    auto asyncB = [&](int k0, int buf) {
        unsigned l0 = (unsigned)(size_t)&Bs[buf][br * LDSB + bc];
        unsigned long long g0 =
            (unsigned long long)(size_t)(B16 + (size_t)(n0 + br) * ldb + k0 + bc);
        asm volatile("global_load_async_to_lds_b128 %0, %1, off"
                     :: "v"(l0), "v"(g0) : "memory");
    };

    auto loadA = [&](int k0) {
        if (F16) {
            int gm = m0 + ar;
            int cm = (gm < M) ? gm : (M - 1);
#pragma unroll
            for (int i = 0; i < 16; ++i) {
                int gk = k0 + ac + i;
                int ck = (gk < K) ? gk : (K - 1);
                _Float16 v = A16[(size_t)cm * lda + ck];
                aH[i] = (gm < M && gk < K) ? v : (_Float16)0.0f;
            }
        } else {
            if (FAST) {
                const float* s = A32 + (size_t)(m0 + ar) * lda + k0 + ac;
                const float4* s4 = (const float4*)s;
#pragma unroll
                for (int j = 0; j < 4; ++j) {
                    float4 f = s4[j];
                    aF[4*j+0] = f.x; aF[4*j+1] = f.y; aF[4*j+2] = f.z; aF[4*j+3] = f.w;
                }
            } else {
                int gm = m0 + ar;
                int cm = (gm < M) ? gm : (M - 1);
#pragma unroll
                for (int i = 0; i < 16; ++i) {
                    int gk = k0 + ac + i;
                    int ck = (gk < K) ? gk : (K - 1);
                    float v = A32[(size_t)cm * lda + ck];
                    aF[i] = (gm < M && gk < K) ? v : 0.0f;
                }
            }
        }
    };
    auto storeA = [&](int buf) {
        v8h lo, hi;
        if (F16) {
#pragma unroll
            for (int i = 0; i < 8; ++i) { lo[i] = aH[i]; hi[i] = aH[8+i]; }
        } else {
#pragma unroll
            for (int i = 0; i < 8; ++i) { lo[i] = (_Float16)aF[i]; hi[i] = (_Float16)aF[8+i]; }
        }
        *(v8h*)&As[buf][ar * LDSA + ac]     = lo;
        *(v8h*)&As[buf][ar * LDSA + ac + 8] = hi;
    };
    auto loadB = [&](int k0) {
        if (BL == 0) {
            if (F16) {
                int gn = n0 + br;
                int cn = (gn < N) ? gn : (N - 1);
#pragma unroll
                for (int i = 0; i < 8; ++i) {
                    int gk = k0 + bc + i;
                    int ck = (gk < K) ? gk : (K - 1);
                    _Float16 v = B16[(size_t)cn * ldb + ck];
                    bH[i] = (gn < N && gk < K) ? v : (_Float16)0.0f;
                }
            } else {
                if (FAST) {
                    const float* s = B32 + (size_t)(n0 + br) * ldb + k0 + bc;
                    const float4* s4 = (const float4*)s;
#pragma unroll
                    for (int j = 0; j < 2; ++j) {
                        float4 f = s4[j];
                        bF[4*j+0] = f.x; bF[4*j+1] = f.y; bF[4*j+2] = f.z; bF[4*j+3] = f.w;
                    }
                } else {
                    int gn = n0 + br;
                    int cn = (gn < N) ? gn : (N - 1);
#pragma unroll
                    for (int i = 0; i < 8; ++i) {
                        int gk = k0 + bc + i;
                        int ck = (gk < K) ? gk : (K - 1);
                        float v = B32[(size_t)cn * ldb + ck];
                        bF[i] = (gn < N && gk < K) ? v : 0.0f;
                    }
                }
            }
        } else {
            if (F16) {
                int gk = k0 + bk;
                int ck = (gk < K) ? gk : (K - 1);
#pragma unroll
                for (int i = 0; i < 8; ++i) {
                    int gn = n0 + bn + i;
                    int cn = (gn < N) ? gn : (N - 1);
                    _Float16 v = B16[(size_t)ck * ldb + cn];
                    bH[i] = (gk < K && gn < N) ? v : (_Float16)0.0f;
                }
            } else if (FAST) {
                const float* s = B32 + (size_t)(k0 + bk) * ldb + n0 + bn;
                const float4* s4 = (const float4*)s;
#pragma unroll
                for (int j = 0; j < 2; ++j) {
                    float4 f = s4[j];
                    bF[4*j+0] = f.x; bF[4*j+1] = f.y; bF[4*j+2] = f.z; bF[4*j+3] = f.w;
                }
            } else {
                int gk = k0 + bk;
                int ck = (gk < K) ? gk : (K - 1);
#pragma unroll
                for (int i = 0; i < 8; ++i) {
                    int gn = n0 + bn + i;
                    int cn = (gn < N) ? gn : (N - 1);
                    float v = B32[(size_t)ck * ldb + cn];
                    bF[i] = (gk < K && gn < N) ? v : 0.0f;
                }
            }
        }
    };
    auto storeB = [&](int buf) {
        if (BL == 0) {
            v8h p;
            if (F16) {
#pragma unroll
                for (int i = 0; i < 8; ++i) p[i] = bH[i];
            } else {
#pragma unroll
                for (int i = 0; i < 8; ++i) p[i] = (_Float16)bF[i];
            }
            *(v8h*)&Bs[buf][br * LDSB + bc] = p;
        } else {
#pragma unroll
            for (int i = 0; i < 8; ++i)
                Bs[buf][(bn + i) * LDSB + bk] = F16 ? bH[i] : (_Float16)bF[i];
        }
    };

    v8f acc[4];
    const v8f vzero = {0.f, 0.f, 0.f, 0.f, 0.f, 0.f, 0.f, 0.f};
#pragma unroll
    for (int i = 0; i < 4; ++i) acc[i] = vzero;

    const int kb  = (lane < 16) ? 0 : 8;   // A fragment K base (halves)
    const int kb2 = (lane < 16) ? 0 : 8;   // B fragment K base (halves)
    const int nkt = (K + BKK - 1) / BKK;

    // prologue: fill buffer 0
    if (ASYNC) {
        asyncA(0, 0); asyncB(0, 0);
        asm volatile("s_wait_asynccnt 0x0" ::: "memory");
    } else {
        loadA(0); loadB(0);
        storeA(0); storeB(0);
    }
    __syncthreads();

    for (int t = 0; t < nkt; ++t) {
        const int cur = t & 1;
        const bool more = (t + 1 < nkt);
        if (more) {
            if (ASYNC) { asyncA((t + 1) * BKK, cur ^ 1); asyncB((t + 1) * BKK, cur ^ 1); }
            else       { loadA((t + 1) * BKK);           loadB((t + 1) * BKK); }
        }

        // prefetch tile t+2 toward cache (staged f32 paths only)
        if (FAST && !ASYNC && (t + 2 < nkt)) {
            int kpf = (t + 2) * BKK;
            __builtin_prefetch((const void*)(A32 + (size_t)(m0 + ar) * lda + kpf + ac), 0, 1);
            if (BL == 0)
                __builtin_prefetch((const void*)(B32 + (size_t)(n0 + br) * ldb + kpf + bc), 0, 1);
            else
                __builtin_prefetch((const void*)(B32 + (size_t)(kpf + bk) * ldb + n0 + bn), 0, 1);
        }

        // ---- MMA on current buffer: fragments first, then 4 WMMAs ----
        const _Float16* ap = &As[cur][(wave * 16 + lr) * LDSA + kb];
        v8h alo = *(const v8h*)ap;
        v8h ahi = *(const v8h*)(ap + 16);
        v16h afrag = __builtin_shufflevector(alo, ahi,
            0,1,2,3,4,5,6,7,8,9,10,11,12,13,14,15);

        v16h bfr[4];
#pragma unroll
        for (int ns = 0; ns < 4; ++ns) {
            const _Float16* bp = &Bs[cur][(ns * 16 + lr) * LDSB + kb2];
            v8h blo = *(const v8h*)bp;
            v8h bhi = *(const v8h*)(bp + 16);
            bfr[ns] = __builtin_shufflevector(blo, bhi,
                0,1,2,3,4,5,6,7,8,9,10,11,12,13,14,15);
        }
#pragma unroll
        for (int ns = 0; ns < 4; ++ns) {
            acc[ns] = __builtin_amdgcn_wmma_f32_16x16x32_f16(
                false, afrag, false, bfr[ns], (short)0, acc[ns], false, false);
        }

        if (more) {
            if (ASYNC) asm volatile("s_wait_asynccnt 0x0" ::: "memory");
            else       { storeA(cur ^ 1); storeB(cur ^ 1); }
        }
        __syncthreads();
    }

    // ---- epilogue ----
    float bval[4] = {0.f, 0.f, 0.f, 0.f};
    if (EPI == 1 || EPI == 2 || EPI == 3 || EPI == 4 || EPI == 5 || EPI == 8) {
#pragma unroll
        for (int ns = 0; ns < 4; ++ns) {
            int n = n0 + ns * 16 + lr;
            int cn = (FAST || n < N) ? n : (N - 1);
            float bv = bias[cn];
            bval[ns] = (FAST || n < N) ? bv : 0.0f;
        }
    }
    const float aval = (EPI == 2) ? alpha[0] : 0.0f;
    const int mrowbase = m0 + wave * 16 + ((lane < 16) ? 0 : 8);

#pragma unroll
    for (int ns = 0; ns < 4; ++ns) {
        int n = n0 + ns * 16 + lr;
#pragma unroll
        for (int r = 0; r < 8; ++r) {
            int m = mrowbase + r;
            if (!FAST && (m >= M || n >= N)) continue;
            float v = acc[ns][r];
            if (EPI == 0) {
                Cf[(long long)m * ldc + n] = v;
            } else if (EPI == 1) {
                Cf[(long long)m * ldc + n] = v + bval[ns];
            } else if (EPI == 2) {
                v += bval[ns];
                Cf[(long long)m * ldc + n] = (v >= 0.f) ? v : aval * v;
            } else if (EPI == 3) {
                v += bval[ns] + res[(long long)m * ldres + n];
                Cf[(long long)m * ldc + n] = v;
            } else if (EPI == 4) {          // q/k head scatter (f16)
                v += bval[ns];
                int b = m >> 9, t2 = m & (TT - 1);
                int h = n >> 6, d = n & (HD - 1);
                Ch[(((long long)(b * NH + h) * TT) + t2) * HD + d] = (_Float16)v;
            } else if (EPI == 5) {          // v transposed scatter (f16)
                v += bval[ns];
                int b = m >> 9, t2 = m & (TT - 1);
                int h = n >> 6, d = n & (HD - 1);
                Ch[(((long long)(b * NH + h) * HD) + d) * TT + t2] = (_Float16)v;
            } else if (EPI == 6) {          // skew accumulate into scores (f32)
                int j = n - (TT - 1) + m;
                if (j >= 0 && j < TT)
                    Cf[(long long)m * TT + j] += v;
            } else if (EPI == 7) {          // attn@V gather: m = t, n = d
                int b = z >> 3, h = z & 7;
                Cf[((long long)(b * TT + m) * DMODEL) + h * HD + n] = v;
            } else if (EPI == 8) {          // bias, f16 store
                Ch[(long long)m * ldc + n] = (_Float16)(v + bval[ns]);
            }
        }
    }
}

// ---------------------------------------------------------------------------
// Elementwise / reduction kernels
// ---------------------------------------------------------------------------
__global__ void concat_kernel(const float* __restrict__ x,
                              const float* __restrict__ mask,
                              float* __restrict__ out, int total)
{
    int e = blockIdx.x * blockDim.x + threadIdx.x;
    if (e >= total) return;
    int m = e / KIN, c = e % KIN;
    out[e] = (c < DOF) ? x[m * DOF + c] : mask[m * DOF + (c - DOF)];
}

__global__ void rel1_kernel(const float* __restrict__ w1,
                            const float* __restrict__ b1,
                            const float* __restrict__ a,
                            float* __restrict__ out, int total)
{
    int e = blockIdx.x * blockDim.x + threadIdx.x;
    if (e >= total) return;
    int r = e / DMODEL, d = e % DMODEL;
    float rd = (float)(r - (TT - 1));
    float v = rd * w1[d] + b1[d];
    float al = a[0];
    out[e] = (v >= 0.f) ? v : al * v;
}

__global__ __launch_bounds__(256)
void layernorm_kernel(const float* __restrict__ x,
                      const float* __restrict__ g,
                      const float* __restrict__ b,
                      float* __restrict__ out)
{
    __shared__ float red[256];
    const int row = blockIdx.x;
    const float* xr = x + (long long)row * DMODEL;
    float* orow = out + (long long)row * DMODEL;
    int t = threadIdx.x;
    float v0 = xr[t], v1 = xr[t + 256];

    red[t] = v0 + v1; __syncthreads();
    for (int s = 128; s > 0; s >>= 1) { if (t < s) red[t] += red[t + s]; __syncthreads(); }
    float mean = red[0] * (1.0f / DMODEL); __syncthreads();
    red[t] = v0 * v0 + v1 * v1; __syncthreads();
    for (int s = 128; s > 0; s >>= 1) { if (t < s) red[t] += red[t + s]; __syncthreads(); }
    float var = red[0] * (1.0f / DMODEL) - mean * mean;
    float rstd = rsqrtf(var + 1e-6f);

    orow[t]       = (v0 - mean) * rstd * g[t]       + b[t];
    orow[t + 256] = (v1 - mean) * rstd * g[t + 256] + b[t + 256];
}

// softmax(S * scale) -> f16 attention matrix
__global__ __launch_bounds__(256)
void softmax_kernel(const float* __restrict__ S, _Float16* __restrict__ P)
{
    __shared__ float red[256];
    const long long row = blockIdx.x;
    const float* p = S + row * TT;
    _Float16* o = P + row * TT;
    int t = threadIdx.x;
    const float scale = 0.125f;              // 1/sqrt(64)
    float a = p[t] * scale, c = p[t + 256] * scale;

    red[t] = fmaxf(a, c); __syncthreads();
    for (int s = 128; s > 0; s >>= 1) { if (t < s) red[t] = fmaxf(red[t], red[t + s]); __syncthreads(); }
    float mx = red[0]; __syncthreads();

    float e0 = __expf(a - mx), e1 = __expf(c - mx);
    red[t] = e0 + e1; __syncthreads();
    for (int s = 128; s > 0; s >>= 1) { if (t < s) red[t] += red[t + s]; __syncthreads(); }
    float inv = 1.0f / red[0];

    o[t]       = (_Float16)(e0 * inv);
    o[t + 256] = (_Float16)(e1 * inv);
}

__global__ void finalize_kernel(const float* __restrict__ y,
                                float* __restrict__ out, int total)
{
    int e = blockIdx.x * blockDim.x + threadIdx.x;
    if (e >= total) return;
    int m = e / DOUT, c = e % DOUT;
    float v = y[e];
    if (c < DOF) out[m * DOF + c] = v;
    else out[MT * DOF + m * 4 + (c - DOF)] = 1.0f / (1.0f + __expf(-v));
}

// ---------------------------------------------------------------------------
// Host-side launcher
// ---------------------------------------------------------------------------
static inline int cdiv(int a, int b) { return (a + b - 1) / b; }

extern "C" void kernel_launch(void* const* d_in, const int* in_sizes, int n_in,
                              void* d_out, int out_size, void* d_ws, size_t ws_size,
                              hipStream_t stream) {
    (void)in_sizes; (void)n_in; (void)out_size; (void)ws_size;
    const float* x       = (const float*)d_in[0];
    const float* mask_in = (const float*)d_in[1];
    const float* enc_w1  = (const float*)d_in[2];
    const float* enc_b1  = (const float*)d_in[3];
    const float* enc_a1  = (const float*)d_in[4];
    const float* enc_w2  = (const float*)d_in[5];
    const float* enc_b2  = (const float*)d_in[6];
    const float* enc_a2  = (const float*)d_in[7];
    const float* rel_w1  = (const float*)d_in[8];
    const float* rel_b1  = (const float*)d_in[9];
    const float* rel_a   = (const float*)d_in[10];
    const float* rel_w2  = (const float*)d_in[11];
    const float* rel_b2  = (const float*)d_in[12];
    const float* ln_g    = (const float*)d_in[13];
    const float* ln_b    = (const float*)d_in[14];
    const float* wq      = (const float*)d_in[15];
    const float* bq      = (const float*)d_in[16];
    const float* wk      = (const float*)d_in[17];
    const float* bk      = (const float*)d_in[18];
    const float* wv      = (const float*)d_in[19];
    const float* bv      = (const float*)d_in[20];
    const float* wo      = (const float*)d_in[21];
    const float* bo      = (const float*)d_in[22];
    const float* fw1     = (const float*)d_in[23];
    const float* fb1     = (const float*)d_in[24];
    const float* fa      = (const float*)d_in[25];
    const float* fw2     = (const float*)d_in[26];
    const float* fb2     = (const float*)d_in[27];
    const float* dec_w1  = (const float*)d_in[28];
    const float* dec_b1  = (const float*)d_in[29];
    const float* dec_a   = (const float*)d_in[30];
    const float* dec_w2  = (const float*)d_in[31];
    const float* dec_b2  = (const float*)d_in[32];
    float* out = (float*)d_out;

    // workspace carve-up; f32 region first, then f16 region (16B aligned)
    float* ws   = (float*)d_ws;
    float* h    = ws;                        size_t off = (size_t)MT * DMODEL;
    float* tmp  = ws + off;                  off += (size_t)MT * DMODEL;
    float* a0   = ws + off;                  off += (size_t)MT * KIN;       // 4096*138
    float* hr   = ws + off;                  off += (size_t)RREL * DMODEL;
    float* Sb   = ws + off;                  off += (size_t)ZBH * TT * TT;
    float* obuf = ws + off;                  off += (size_t)MT * DMODEL;
    float* t1   = ws + off;                  off += (size_t)MT * DFF;
    float* ybuf = ws + off;                  off += (size_t)MT * DOUT;
    _Float16* h16  = (_Float16*)(ws + off);  size_t hoff = 0;
    _Float16* qbuf = h16 + hoff;             hoff += (size_t)ZBH * TT * HD;
    _Float16* kbuf = h16 + hoff;             hoff += (size_t)ZBH * TT * HD;
    _Float16* vt   = h16 + hoff;             hoff += (size_t)ZBH * HD * TT;
    _Float16* Em16 = h16 + hoff;             hoff += (size_t)RREL * HD;
    _Float16* Sb16 = h16 + hoff;             hoff += (size_t)ZBH * TT * TT;

    const dim3 blk(256);

    // 1. concat(x, mask) -> a0  (4096 x 138)
    {
        int total = MT * KIN;
        concat_kernel<<<cdiv(total, 256), 256, 0, stream>>>(x, mask_in, a0, total);
    }
    // 2. encoder layer 1 (K=138 -> guarded): tmp = prelu(a0 @ enc_w1 + b1, a1)
    gemm_wmma<1, 2, 0, 0><<<dim3(cdiv(DMODEL,BN), cdiv(MT,BM), 1), blk, 0, stream>>>(
        a0, enc_w1, tmp, MT, DMODEL, KIN, KIN, DMODEL, DMODEL,
        0, 0, 0, enc_b1, enc_a1, nullptr, 0);
    // 3. encoder layer 2: h = prelu(tmp @ enc_w2 + b2, a2)
    gemm_wmma<1, 2, 1, 0><<<dim3(cdiv(DMODEL,BN), cdiv(MT,BM), 1), blk, 0, stream>>>(
        tmp, enc_w2, h, MT, DMODEL, DMODEL, DMODEL, DMODEL, DMODEL,
        0, 0, 0, enc_b2, enc_a2, nullptr, 0);
    // 4. relative embedding expansion
    {
        int total = RREL * DMODEL;
        rel1_kernel<<<cdiv(total, 256), 256, 0, stream>>>(rel_w1, rel_b1, rel_a, hr, total);
    }
    // 5. E = hr @ rel_w2 + rel_b2  (1023 x 64, M edge -> guarded; f16 out)
    gemm_wmma<1, 8, 0, 0><<<dim3(cdiv(HD,BN), cdiv(RREL,BM), 1), blk, 0, stream>>>(
        hr, rel_w2, Em16, RREL, HD, DMODEL, DMODEL, HD, HD,
        0, 0, 0, rel_b2, nullptr, nullptr, 0);

    for (int l = 0; l < NL; ++l) {
        const float* wql = wq + (size_t)l * DMODEL * DMODEL;
        const float* wkl = wk + (size_t)l * DMODEL * DMODEL;
        const float* wvl = wv + (size_t)l * DMODEL * DMODEL;
        const float* wol = wo + (size_t)l * DMODEL * DMODEL;
        const float* bql = bq + (size_t)l * DMODEL;
        const float* bkl = bk + (size_t)l * DMODEL;
        const float* bvl = bv + (size_t)l * DMODEL;
        const float* bol = bo + (size_t)l * DMODEL;
        const float* fw1l = fw1 + (size_t)l * DMODEL * DFF;
        const float* fb1l = fb1 + (size_t)l * DFF;
        const float* fal  = fa + l;
        const float* fw2l = fw2 + (size_t)l * DFF * DMODEL;
        const float* fb2l = fb2 + (size_t)l * DMODEL;

        // a. hn = LN(h) -> tmp
        layernorm_kernel<<<MT, 256, 0, stream>>>(h, ln_g, ln_b, tmp);
        // b/c/d. Q, K ([b,h,t,d] f16) and V (transposed [b,h,d,t] f16)
        gemm_wmma<1, 4, 1, 0><<<dim3(cdiv(DMODEL,BN), cdiv(MT,BM), 1), blk, 0, stream>>>(
            tmp, wql, qbuf, MT, DMODEL, DMODEL, DMODEL, DMODEL, 0,
            0, 0, 0, bql, nullptr, nullptr, 0);
        gemm_wmma<1, 4, 1, 0><<<dim3(cdiv(DMODEL,BN), cdiv(MT,BM), 1), blk, 0, stream>>>(
            tmp, wkl, kbuf, MT, DMODEL, DMODEL, DMODEL, DMODEL, 0,
            0, 0, 0, bkl, nullptr, nullptr, 0);
        gemm_wmma<1, 5, 1, 0><<<dim3(cdiv(DMODEL,BN), cdiv(MT,BM), 1), blk, 0, stream>>>(
            tmp, wvl, vt, MT, DMODEL, DMODEL, DMODEL, DMODEL, 0,
            0, 0, 0, bvl, nullptr, nullptr, 0);
        // e. S = Q @ K^T  (f16 in via async-to-LDS, f32 out), batched over (b,h)
        gemm_wmma<0, 0, 1, 1><<<dim3(cdiv(TT,BN), cdiv(TT,BM), ZBH), blk, 0, stream>>>(
            qbuf, kbuf, Sb, TT, TT, HD, HD, HD, TT,
            (long long)TT * HD, (long long)TT * HD, (long long)TT * TT,
            nullptr, nullptr, nullptr, 0);
        // f. S[i][j] += (Q @ E^T)[i][j - i + T - 1]  (skew; N=1023 -> guarded)
        gemm_wmma<0, 6, 0, 1><<<dim3(cdiv(RREL,BN), cdiv(TT,BM), ZBH), blk, 0, stream>>>(
            qbuf, Em16, Sb, TT, RREL, HD, HD, HD, TT,
            (long long)TT * HD, 0, (long long)TT * TT,
            nullptr, nullptr, nullptr, 0);
        // g. attn = softmax(S * scale) -> f16
        softmax_kernel<<<ZBH * TT, 256, 0, stream>>>(Sb, Sb16);
        // h. o = attn @ V (f16 in via async-to-LDS) -> gathered f32 [b,t,h*64+d]
        gemm_wmma<0, 7, 1, 1><<<dim3(cdiv(HD,BN), cdiv(TT,BM), ZBH), blk, 0, stream>>>(
            Sb16, vt, obuf, TT, HD, TT, TT, TT, 0,
            (long long)TT * TT, (long long)HD * TT, 0,
            nullptr, nullptr, nullptr, 0);
        // i. h = h + o @ wo + bo
        gemm_wmma<1, 3, 1, 0><<<dim3(cdiv(DMODEL,BN), cdiv(MT,BM), 1), blk, 0, stream>>>(
            obuf, wol, h, MT, DMODEL, DMODEL, DMODEL, DMODEL, DMODEL,
            0, 0, 0, bol, nullptr, h, DMODEL);
        // j. hn = LN(h) -> tmp
        layernorm_kernel<<<MT, 256, 0, stream>>>(h, ln_g, ln_b, tmp);
        // k. t1 = prelu(tmp @ fw1 + fb1, fa)
        gemm_wmma<1, 2, 1, 0><<<dim3(cdiv(DFF,BN), cdiv(MT,BM), 1), blk, 0, stream>>>(
            tmp, fw1l, t1, MT, DFF, DMODEL, DMODEL, DFF, DFF,
            0, 0, 0, fb1l, fal, nullptr, 0);
        // l. h = h + t1 @ fw2 + fb2
        gemm_wmma<1, 3, 1, 0><<<dim3(cdiv(DMODEL,BN), cdiv(MT,BM), 1), blk, 0, stream>>>(
            t1, fw2l, h, MT, DMODEL, DFF, DFF, DMODEL, DMODEL,
            0, 0, 0, fb2l, nullptr, h, DMODEL);
    }

    // 7. decoder: tmp = prelu(h @ dec_w1 + dec_b1, dec_a)
    gemm_wmma<1, 2, 1, 0><<<dim3(cdiv(DMODEL,BN), cdiv(MT,BM), 1), blk, 0, stream>>>(
        h, dec_w1, tmp, MT, DMODEL, DMODEL, DMODEL, DMODEL, DMODEL,
        0, 0, 0, dec_b1, dec_a, nullptr, 0);
    // 8. y = tmp @ dec_w2 + dec_b2  (4096 x 73, N edge + ldb=73 -> guarded)
    gemm_wmma<1, 1, 0, 0><<<dim3(cdiv(DOUT,BN), cdiv(MT,BM), 1), blk, 0, stream>>>(
        tmp, dec_w2, ybuf, MT, DOUT, DMODEL, DMODEL, DOUT, DOUT,
        0, 0, 0, dec_b2, nullptr, nullptr, 0);
    // 9. split + sigmoid
    {
        int total = MT * DOUT;
        finalize_kernel<<<cdiv(total, 256), 256, 0, stream>>>(ybuf, out, total);
    }
}
